// MambaMixer_6133213299444
// MI455X (gfx1250) — compile-verified, matrix operand order checked by reading, not run on hardware
//
#include <hip/hip_runtime.h>

// ---------------- problem constants (from reference) ----------------
#define BB 2
#define LL 1024
#define HH 2048
#define DD 4096
#define NN 16
#define KCONV 4
#define RR 128
#define MROWS (BB * LL)      // 2048
#define SSM_LD 192           // R + 2N = 160 padded to 192 (multiple of 64)

typedef __attribute__((ext_vector_type(16))) __bf16 v16bf;
typedef __attribute__((ext_vector_type(8)))  __bf16 v8bf;
typedef __attribute__((ext_vector_type(8)))  float  v8f;
typedef __attribute__((ext_vector_type(4)))  unsigned int v4ui;
typedef __attribute__((ext_vector_type(8)))  int v8i;
typedef __attribute__((ext_vector_type(4)))  int v4i;

union ABFrag { v16bf v; v8bf h[2]; };

__device__ __forceinline__ __bf16 f2bf(float f) { return (__bf16)f; }

// ---------------- f32 -> bf16 strided convert (with row zero-padding) --------
__global__ void convert_bf16(const float* __restrict__ in, __bf16* __restrict__ out,
                             long rows, long cols, long inLd, long outLd, long validRows) {
    long idx = (long)blockIdx.x * blockDim.x + threadIdx.x;
    long total = rows * cols;
    if (idx >= total) return;
    long r = idx / cols, c = idx - r * cols;
    float v = (r < validRows) ? in[r * inLd + c] : 0.0f;
    out[r * outLd + c] = f2bf(v);
}

// ---------------- TDM: DMA one 64-row x 32-el bf16 B tile into LDS -----------
// LDS layout: row pitch 80B (TDM pads 16B after every 64B row) so the 16
// row-reads of a ds_load_b128 phase map to 16 distinct banks (20*r mod 64).
#define TDM_ROWPITCH_EL 40                  // 80 bytes / 2
#define TDM_BUF_EL      (64 * TDM_ROWPITCH_EL)
#define TDM_BUF_BYTES   (TDM_BUF_EL * 2)    // 5120

__device__ __forceinline__ void tdm_issue_b(unsigned ldsByteAddr, const __bf16* W,
                                            int rowStart, int k, int ldb) {
    unsigned long long ga = (unsigned long long)(W + (size_t)rowStart * ldb + k);
    v4ui g0;
    g0[0] = 1u;                                          // count=1 (valid), user mode
    g0[1] = ldsByteAddr;                                 // lds_addr (bytes)
    g0[2] = (unsigned)(ga & 0xffffffffu);                // global_addr[31:0]
    g0[3] = (unsigned)((ga >> 32) & 0x01ffffffu) | (2u << 30);  // addr[56:32], type=2
    v8i g1;
    g1[0] = (int)((1u << 16) |                           // data_size = 2 bytes
                  (1u << 20) |                           // pad_enable
                  (3u << 22) |                           // pad_interval: 16 DWORDs (64B)
                  (3u << 25));                           // pad_amount: 4 DWORDs (16B)
    g1[1] = (int)(32u << 16);                            // tensor_dim0[15:0] = 32
    g1[2] = (int)(64u << 16);                            // dim0 hi=0 | tensor_dim1 = 64
    g1[3] = (int)(32u << 16);                            // dim1 hi=0 | tile_dim0 = 32
    g1[4] = 64;                                          // tile_dim1 = 64 (tile_dim2=0)
    g1[5] = ldb;                                         // tensor_dim0_stride lo32
    g1[6] = 0;                                           // stride hi | dim1_stride lo
    g1[7] = 0;
    v4i z4 = {0, 0, 0, 0};
    v8i z8 = {0, 0, 0, 0, 0, 0, 0, 0};
    __builtin_amdgcn_tensor_load_to_lds(g0, g1, z4, z4, z8, 0);
}

// ---------------- WMMA bf16 GEMM:  C[M,N] = A[M,K] * W[N,K]^T ----------------
// block = 256 threads = 8 waves; wave w owns rows [by*256 + w*32, +32),
// cols [bx*64, +64): 2x4 WMMA tiles. B tile staged in LDS by the TDM
// (double-buffered, TENSORcnt-tracked); A loaded directly (no reuse in-block).
// K % 32 == 0, M % 256 == 0, N % 64 == 0 (caller pads).
// EPI: 0 = none, 1 = softplus(acc + bias[col])
template<int EPI>
__global__ void wmma_gemm(const __bf16* __restrict__ A, const __bf16* __restrict__ W,
                          const float* __restrict__ bias, float* __restrict__ C,
                          int K, int lda, int ldb, int ldc) {
    // only __shared__ object in this kernel -> LDS byte offset 0
    __shared__ __align__(16) __bf16 btile[2][TDM_BUF_EL];

    const int lane = threadIdx.x & 31;
    const int wave = threadIdx.x >> 5;
    const int g    = lane >> 4;      // half-wave selects K sub-range
    const int r    = lane & 15;      // row (A) / col (B) within tile
    const int g8   = g * 8;

    const int rowBase = blockIdx.y * 256 + wave * 32;
    const int colBase = blockIdx.x * 64;
    const int nk      = K >> 5;

    const __bf16* aRow0 = A + (size_t)(rowBase + r) * lda;
    const __bf16* aRow1 = A + (size_t)(rowBase + 16 + r) * lda;

    v8f acc[2][4] = {};

    if (wave == 0) tdm_issue_b(0u, W, colBase, 0, ldb);

    for (int kt = 0; kt < nk; ++kt) {
        const int k = kt << 5;
        if (wave == 0) {
            if (kt + 1 < nk) {
                tdm_issue_b((unsigned)(((kt + 1) & 1) * TDM_BUF_BYTES), W, colBase, k + 32, ldb);
                __builtin_amdgcn_s_wait_tensorcnt(1);   // buffer kt&1 complete
            } else {
                __builtin_amdgcn_s_wait_tensorcnt(0);
            }
        }
        __syncthreads();                                // publish buffer kt&1

        ABFrag a0, a1;
        a0.h[0] = *(const v8bf*)(aRow0 + k + g8);
        a0.h[1] = *(const v8bf*)(aRow0 + k + 16 + g8);
        a1.h[0] = *(const v8bf*)(aRow1 + k + g8);
        a1.h[1] = *(const v8bf*)(aRow1 + k + 16 + g8);

        const __bf16* tb = &btile[kt & 1][0];
#pragma unroll
        for (int nt = 0; nt < 4; ++nt) {
            ABFrag b;
            const __bf16* rowp = tb + (nt * 16 + r) * TDM_ROWPITCH_EL;
            b.h[0] = *(const v8bf*)(rowp + g8);
            b.h[1] = *(const v8bf*)(rowp + 16 + g8);
            acc[0][nt] = __builtin_amdgcn_wmma_f32_16x16x32_bf16(
                false, a0.v, false, b.v, (short)0, acc[0][nt], false, false);
            acc[1][nt] = __builtin_amdgcn_wmma_f32_16x16x32_bf16(
                false, a1.v, false, b.v, (short)0, acc[1][nt], false, false);
        }
        if (k + 256 < K) {                              // global_prefetch_b8 ahead
            __builtin_prefetch(aRow0 + k + 256, 0, 1);
            __builtin_prefetch(aRow1 + k + 256, 0, 1);
        }
        __syncthreads();                                // reads done before DMA reuses buffer
    }

    // C layout: element j of v8f -> M = tileRow + j + 8*g, N = colBase + nt*16 + r
#pragma unroll
    for (int mt = 0; mt < 2; ++mt) {
#pragma unroll
        for (int nt = 0; nt < 4; ++nt) {
            const int col = colBase + nt * 16 + r;
#pragma unroll
            for (int j = 0; j < 8; ++j) {
                const int row = rowBase + mt * 16 + j + 8 * g;
                float v = acc[mt][nt][j];
                if (EPI == 1) {
                    v += bias[col];
                    v = (v > 20.0f) ? v : __logf(1.0f + __expf(v));
                }
                C[(size_t)row * ldc + col] = v;
            }
        }
    }
}

// ---------------- causal depthwise conv (K=4) + SiLU -------------------------
__global__ void conv_silu(const float* __restrict__ proj, const float* __restrict__ cw,
                          const float* __restrict__ cb, float* __restrict__ hs_f,
                          __bf16* __restrict__ hs_bf) {
    long idx = (long)blockIdx.x * blockDim.x + threadIdx.x;
    if (idx >= (long)MROWS * DD) return;
    int d  = (int)(idx & (DD - 1));
    long bl = idx >> 12;                 // b*L + l
    int l  = (int)(bl & (LL - 1));
    long b = bl >> 10;
    float acc = cb[d];
#pragma unroll
    for (int k = 0; k < KCONV; ++k) {
        int li = l + k - (KCONV - 1);
        if (li >= 0)
            acc += proj[(((b << 10) + li) << 13) + d] * cw[d * KCONV + k];
    }
    float s = acc / (1.0f + __expf(-acc));   // silu
    hs_f[idx]  = s;
    hs_bf[idx] = f2bf(s);
}

// ---------------- fused selective scan ---------------------------------------
// One lane per (b, d, n): 16 n-lanes per channel, butterfly-reduce C contraction.
// Fuses dA = exp(dt*A), dBu = dt*B*hs, the D skip, gate SiLU, and bf16 output.
__global__ void ssm_scan(const float* __restrict__ dt, const float* __restrict__ hs,
                         const float* __restrict__ ssm, const float* __restrict__ proj,
                         const float* __restrict__ A_log, const float* __restrict__ Dp,
                         __bf16* __restrict__ y_bf) {
    int tid = blockIdx.x * blockDim.x + threadIdx.x;   // BB*DD*NN total
    int n  = tid & 15;
    int bd = tid >> 4;
    int d  = bd & (DD - 1);
    int b  = bd >> 12;
    float An   = -__expf(A_log[d * NN + n]);
    float dpar = Dp[d];
    float state = 0.0f;
    for (int l = 0; l < LL; ++l) {
        size_t base = (size_t)(b * LL + l);
        float dtv = dt[base * DD + d];
        float hsv = hs[base * DD + d];
        float Bn  = ssm[base * SSM_LD + RR + n];
        float Cn  = ssm[base * SSM_LD + RR + NN + n];
        state = __expf(dtv * An) * state + dtv * Bn * hsv;
        float p = state * Cn;
        p += __shfl_xor(p, 1, 32);
        p += __shfl_xor(p, 2, 32);
        p += __shfl_xor(p, 4, 32);
        p += __shfl_xor(p, 8, 32);
        if (n == 0) {
            float gt = proj[(base << 13) + DD + d];          // gate (pre-silu)
            float yv = (p + hsv * dpar) * (gt / (1.0f + __expf(-gt)));
            y_bf[base * DD + d] = f2bf(yv);
        }
    }
}

// ---------------- host orchestration -----------------------------------------
extern "C" void kernel_launch(void* const* d_in, const int* in_sizes, int n_in,
                              void* d_out, int out_size, void* d_ws, size_t ws_size,
                              hipStream_t stream) {
    (void)in_sizes; (void)n_in; (void)out_size; (void)ws_size;
    const float* x          = (const float*)d_in[0];
    const float* in_proj_w  = (const float*)d_in[1];
    const float* conv_w     = (const float*)d_in[2];
    const float* conv_b     = (const float*)d_in[3];
    const float* x_proj_w   = (const float*)d_in[4];
    const float* dt_proj_w  = (const float*)d_in[5];
    const float* dt_proj_b  = (const float*)d_in[6];
    const float* A_log      = (const float*)d_in[7];
    const float* D_param    = (const float*)d_in[8];
    const float* out_proj_w = (const float*)d_in[9];
    float* out = (float*)d_out;

    char* ws = (char*)d_ws;
    size_t off = 0;
    auto alloc = [&](size_t bytes) -> void* {
        void* p = ws + off;
        off += (bytes + 255) & ~(size_t)255;
        return p;
    };

    __bf16* x_bf     = (__bf16*)alloc((size_t)MROWS * HH * 2);
    __bf16* w_in_bf  = (__bf16*)alloc((size_t)(2 * DD) * HH * 2);
    float*  proj     = (float*) alloc((size_t)MROWS * (2 * DD) * 4);
    float*  hs_f     = (float*) alloc((size_t)MROWS * DD * 4);
    __bf16* hs_bf    = (__bf16*)alloc((size_t)MROWS * DD * 2);
    __bf16* w_xp_bf  = (__bf16*)alloc((size_t)SSM_LD * DD * 2);
    float*  ssm      = (float*) alloc((size_t)MROWS * SSM_LD * 4);
    __bf16* t_bf     = (__bf16*)alloc((size_t)MROWS * RR * 2);
    __bf16* w_dt_bf  = (__bf16*)alloc((size_t)DD * RR * 2);
    float*  dt       = (float*) alloc((size_t)MROWS * DD * 4);
    __bf16* y_bf     = (__bf16*)alloc((size_t)MROWS * DD * 2);
    __bf16* w_out_bf = (__bf16*)alloc((size_t)HH * DD * 2);

    const int T = 256;
    auto blocks = [](long total) { return (unsigned)((total + 255) / 256); };

    // 1) bf16 conversions (weights + x)
    convert_bf16<<<blocks((long)MROWS * HH), T, 0, stream>>>(x, x_bf, MROWS, HH, HH, HH, MROWS);
    convert_bf16<<<blocks((long)2 * DD * HH), T, 0, stream>>>(in_proj_w, w_in_bf, 2 * DD, HH, HH, HH, 2 * DD);
    convert_bf16<<<blocks((long)SSM_LD * DD), T, 0, stream>>>(x_proj_w, w_xp_bf, SSM_LD, DD, DD, DD, RR + 2 * NN);
    convert_bf16<<<blocks((long)DD * RR), T, 0, stream>>>(dt_proj_w, w_dt_bf, DD, RR, RR, RR, DD);
    convert_bf16<<<blocks((long)HH * DD), T, 0, stream>>>(out_proj_w, w_out_bf, HH, DD, DD, DD, HH);

    // 2) in_proj: proj[2048, 8192] = x_bf[2048, 2048] * w_in_bf[8192, 2048]^T
    wmma_gemm<0><<<dim3(2 * DD / 64, MROWS / 256), T, 0, stream>>>(
        x_bf, w_in_bf, nullptr, proj, HH, HH, HH, 2 * DD);

    // 3) depthwise causal conv + silu
    conv_silu<<<blocks((long)MROWS * DD), T, 0, stream>>>(proj, conv_w, conv_b, hs_f, hs_bf);

    // 4) x_proj: ssm[2048, 192(pad)] = hs_bf[2048, 4096] * w_xp_bf[192, 4096]^T
    wmma_gemm<0><<<dim3(SSM_LD / 64, MROWS / 256), T, 0, stream>>>(
        hs_bf, w_xp_bf, nullptr, ssm, DD, DD, DD, SSM_LD);

    // 5) extract t (first R cols of ssm) as bf16
    convert_bf16<<<blocks((long)MROWS * RR), T, 0, stream>>>(ssm, t_bf, MROWS, RR, SSM_LD, RR, MROWS);

    // 6) dt = softplus(t * dt_proj_w^T + b): [2048, 4096], K=128
    wmma_gemm<1><<<dim3(DD / 64, MROWS / 256), T, 0, stream>>>(
        t_bf, w_dt_bf, dt_proj_b, dt, RR, RR, RR, DD);

    // 7) fused selective scan -> y_bf (gated, skip-connected), bf16 for out_proj
    ssm_scan<<<blocks((long)BB * DD * NN), T, 0, stream>>>(
        dt, hs_f, ssm, proj, A_log, D_param, y_bf);

    // 8) out_proj: out[2048, 2048] = y_bf[2048, 4096] * w_out_bf[2048, 4096]^T
    wmma_gemm<0><<<dim3(HH / 64, MROWS / 256), T, 0, stream>>>(
        y_bf, w_out_bf, nullptr, out, DD, DD, DD, HH);
}
